// OctreeAttention_89172111000145
// MI455X (gfx1250) — compile-verified, matrix operand order checked by reading, not run on hardware
//
#include <hip/hip_runtime.h>

// ---------------------------------------------------------------------------
// Octree windowed attention for MI455X (gfx1250), bf16 WMMA pipeline.
// ---------------------------------------------------------------------------

typedef __bf16 bf16_t;
typedef __attribute__((ext_vector_type(16))) __bf16 v16bf;
typedef __attribute__((ext_vector_type(8)))  float  v8f;

#define N_PTS    65536
#define C_DIM    512
#define NH       8
#define HDIM     64
#define KW       24
#define PAD_N    12
#define NA_PTS   65568
#define NWIN     2732
#define QKV_COLS 1536

__device__ __forceinline__ bf16_t f2bf(float x) { return (bf16_t)x; }

__device__ __forceinline__ v8f v8f_zero() {
  v8f z = {0.f,0.f,0.f,0.f,0.f,0.f,0.f,0.f};
  return z;
}

// K-element index for 16-bit WMMA A/B fragments (CDNA5 ISA 7.12.2):
// lane holds row/col = lane&15; VGPR j pair covers K = 2j + (j>=4?8:0) + (lane>=16?8:0)
__device__ __forceinline__ int frag_kbase(int j, int hi) {
  return 2 * j + ((j >> 2) << 3) + (hi << 3);
}

// Load a 16x32 bf16 fragment from a row-major LDS region; rowptr = &row[0].
__device__ __forceinline__ v16bf load_frag(const bf16_t* rowptr, int lane) {
  int hi = lane >> 4;
  v16bf f;
#pragma unroll
  for (int j = 0; j < 8; ++j) {
    int k = frag_kbase(j, hi);
    f[2 * j]     = rowptr[k];
    f[2 * j + 1] = rowptr[k + 1];
  }
  return f;
}

// Same, but rows >= rowlim read as zero (tile padding 24 -> 32).
__device__ __forceinline__ v16bf load_frag_clamped(const bf16_t* base, int row, int stride,
                                                   int koff, int lane, int rowlim) {
  int hi = lane >> 4;
  bool ok = row < rowlim;
  const bf16_t* rp = base + (size_t)(ok ? row : 0) * stride + koff;
  v16bf f;
#pragma unroll
  for (int j = 0; j < 8; ++j) {
    int k = frag_kbase(j, hi);
    f[2 * j]     = ok ? rp[k]     : f2bf(0.f);
    f[2 * j + 1] = ok ? rp[k + 1] : f2bf(0.f);
  }
  return f;
}

// B fragment for P@V: lane = output column, elements run along kv rows.
__device__ __forceinline__ v16bf load_frag_v(const bf16_t* vbase, int stride, int col, int lane) {
  int hi = lane >> 4;
  v16bf f;
#pragma unroll
  for (int j = 0; j < 8; ++j) {
    int kv = frag_kbase(j, hi);
    f[2 * j]     = (kv < KW)     ? vbase[(size_t)kv * stride + col]       : f2bf(0.f);
    f[2 * j + 1] = (kv + 1 < KW) ? vbase[(size_t)(kv + 1) * stride + col] : f2bf(0.f);
  }
  return f;
}

// ---------------------------------------------------------------------------
// Kernel 0: fill pad rows of padded qkv buffer (head = -1, tail = 0).
// ---------------------------------------------------------------------------
__global__ void pad_kernel(bf16_t* __restrict__ qkvw) {
  int i = blockIdx.x * 256 + threadIdx.x;
  if (i >= 32 * QKV_COLS) return;
  int r = i / QKV_COLS, c = i % QKV_COLS;
  int a;
  float v;
  if (r < PAD_N) { a = r; v = -1.0f; }
  else           { a = (N_PTS + PAD_N) + (r - PAD_N); v = 0.0f; }
  qkvw[(size_t)a * QKV_COLS + c] = f2bf(v);
}

// ---------------------------------------------------------------------------
// Kernel 1: qkv = data @ Wqkv + bqkv, RoPE on q/k, store bf16 at row a=n+PAD.
// 128x128 tile / workgroup, 8 waves each computing 32x64 (2x4 WMMA tiles).
// ---------------------------------------------------------------------------
__global__ __launch_bounds__(256) void qkv_rope_kernel(
    const float* __restrict__ data, const float* __restrict__ xyz,
    const float* __restrict__ Wqkv, const float* __restrict__ bqkv,
    const float* __restrict__ rope_freqs, bf16_t* __restrict__ qkvw) {
  __shared__ __align__(16) bf16_t As[128][34];   // row-major [m][k]
  __shared__ __align__(16) bf16_t Bs[128][34];   // transposed [n][k]

  const int tid = threadIdx.x, lane = tid & 31, wid = tid >> 5;
  const int wr = wid >> 1, wc = wid & 1;
  const int m0 = blockIdx.x * 128, n0 = blockIdx.y * 128;
  const int wm = wr * 32, wn = wc * 64;
  const int cl = lane & 15, hi = lane >> 4;

  v8f acc[2][4];
#pragma unroll
  for (int i = 0; i < 2; ++i)
#pragma unroll
    for (int j = 0; j < 4; ++j) acc[i][j] = v8f_zero();

  for (int kc = 0; kc < C_DIM; kc += 32) {
#pragma unroll
    for (int i = 0; i < 4; ++i) {          // A stage: 1024 float4
      int idx = tid * 4 + i;
      int r = idx >> 3, c = idx & 7;
      float4 vv = *(const float4*)(data + (size_t)(m0 + r) * C_DIM + kc + c * 4);
      As[r][c * 4 + 0] = f2bf(vv.x); As[r][c * 4 + 1] = f2bf(vv.y);
      As[r][c * 4 + 2] = f2bf(vv.z); As[r][c * 4 + 3] = f2bf(vv.w);
    }
#pragma unroll
    for (int i = 0; i < 4; ++i) {          // B stage (transposed)
      int idx = tid * 4 + i;
      int kr = idx >> 5, cg = idx & 31;
      float4 vv = *(const float4*)(Wqkv + (size_t)(kc + kr) * QKV_COLS + n0 + cg * 4);
      Bs[cg * 4 + 0][kr] = f2bf(vv.x); Bs[cg * 4 + 1][kr] = f2bf(vv.y);
      Bs[cg * 4 + 2][kr] = f2bf(vv.z); Bs[cg * 4 + 3][kr] = f2bf(vv.w);
    }
    if (kc + 32 < C_DIM) {
      __builtin_prefetch(data + (size_t)(m0 + (tid >> 1)) * C_DIM + kc + 32, 0, 1);
      __builtin_prefetch(Wqkv + (size_t)(kc + 32 + (tid >> 3)) * QKV_COLS + n0, 0, 1);
    }
    __syncthreads();

    v16bf af[2], bfr[4];
#pragma unroll
    for (int mt = 0; mt < 2; ++mt)
      af[mt] = load_frag(&As[wm + mt * 16 + cl][0], lane);
#pragma unroll
    for (int nt = 0; nt < 4; ++nt)
      bfr[nt] = load_frag(&Bs[wn + nt * 16 + cl][0], lane);
#pragma unroll
    for (int mt = 0; mt < 2; ++mt)
#pragma unroll
      for (int nt = 0; nt < 4; ++nt)
        acc[mt][nt] = __builtin_amdgcn_wmma_f32_16x16x32_bf16(
            false, af[mt], false, bfr[nt], (short)0, acc[mt][nt], false, false);
    __syncthreads();
  }

  // Epilogue: bias + RoPE (wave's 64 cols live in exactly one (s, head) slice).
  const int colbase = n0 + wn;
  const int s = colbase / C_DIM;
  const int h = (colbase % C_DIM) / HDIM;
  float bias[4];
#pragma unroll
  for (int nt = 0; nt < 4; ++nt) bias[nt] = bqkv[colbase + nt * 16 + cl];

#pragma unroll
  for (int mt = 0; mt < 2; ++mt) {
#pragma unroll
    for (int vr = 0; vr < 8; ++vr) {
      int m = m0 + wm + mt * 16 + vr + hi * 8;
      size_t dst = (size_t)(m + PAD_N) * QKV_COLS + colbase;
      if (s < 2) {   // q or k: rotate pairs (p, p+32)
        float x0 = xyz[m * 3 + 0], x1 = xyz[m * 3 + 1], x2 = xyz[m * 3 + 2];
#pragma unroll
        for (int ni = 0; ni < 2; ++ni) {
          int p = ni * 16 + cl;
          const float* fp = rope_freqs + (size_t)(h * 32 + p) * 3;
          float th = x0 * fp[0] + x1 * fp[1] + x2 * fp[2];
          float cs = __cosf(th), sn = __sinf(th);
          float t1 = acc[mt][ni][vr]     + bias[ni];
          float t2 = acc[mt][ni + 2][vr] + bias[ni + 2];
          qkvw[dst + p]      = f2bf(t1 * cs - t2 * sn);
          qkvw[dst + p + 32] = f2bf(t1 * sn + t2 * cs);
        }
      } else {       // v: bias only
#pragma unroll
        for (int nt = 0; nt < 4; ++nt)
          qkvw[dst + nt * 16 + cl] = f2bf(acc[mt][nt][vr] + bias[nt]);
      }
    }
  }
}

// ---------------------------------------------------------------------------
// Kernel 2: windowed attention. One workgroup = (window, 4 heads); one wave
// per head does S = qk^T (8 WMMA), masked softmax in registers, O = PV (8 WMMA).
// ---------------------------------------------------------------------------
__global__ __launch_bounds__(128) void attn_kernel(
    const bf16_t* __restrict__ qkvw, const float* __restrict__ mask,
    bf16_t* __restrict__ aout) {
  __shared__ __align__(16) bf16_t sQKV[KW][768];   // [row][s*256 + headlocal*64 + hd]
  __shared__ float sMask[KW * KW];
  __shared__ __align__(16) bf16_t sP[4][32 * 32];  // per-wave softmax staging

  const int w  = blockIdx.x;
  const int hb = blockIdx.y * 4;                   // head base for this block
  const int g = w >> 2, d = w & 3;
  const int tid = threadIdx.x, lane = tid & 31, wid = tid >> 5;
  const int h = hb + wid;
  const int cl = lane & 15, hi = lane >> 4;

  // Gather this window's 24 dilated rows (q/k/v slices of 4 heads).
  for (int i = tid; i < KW * 96; i += 128) {
    int p = i / 96, rem = i % 96;
    int s = rem >> 5, off = rem & 31;
    int a = g * 96 + p * 4 + d;
    const uint4* src = (const uint4*)(qkvw + (size_t)a * QKV_COLS + s * C_DIM + hb * HDIM);
    ((uint4*)&sQKV[p][s * 256])[off] = src[off];
  }
  for (int i = tid; i < KW * KW; i += 128) sMask[i] = mask[(size_t)w * (KW * KW) + i];
  __syncthreads();

  const bf16_t* qp = &sQKV[0][wid * 64];
  const bf16_t* kp = &sQKV[0][256 + wid * 64];
  const bf16_t* vp = &sQKV[0][512 + wid * 64];
  const int SQ = 768;

  // S = q @ k^T  (24x24 padded to 32x32; K = 64 in 2 chunks)
  v8f sacc[2][2];
#pragma unroll
  for (int i = 0; i < 2; ++i)
#pragma unroll
    for (int j = 0; j < 2; ++j) sacc[i][j] = v8f_zero();
#pragma unroll
  for (int kc = 0; kc < 2; ++kc) {
    v16bf aq[2], bk[2];
#pragma unroll
    for (int mt = 0; mt < 2; ++mt)
      aq[mt] = load_frag_clamped(qp, mt * 16 + cl, SQ, kc * 32, lane, KW);
#pragma unroll
    for (int nt = 0; nt < 2; ++nt)
      bk[nt] = load_frag_clamped(kp, nt * 16 + cl, SQ, kc * 32, lane, KW);
#pragma unroll
    for (int mt = 0; mt < 2; ++mt)
#pragma unroll
      for (int nt = 0; nt < 2; ++nt)
        sacc[mt][nt] = __builtin_amdgcn_wmma_f32_16x16x32_bf16(
            false, aq[mt], false, bk[nt], (short)0, sacc[mt][nt], false, false);
  }

  // Masked softmax per query row (cols spread across 16 lanes x 2 tiles).
  const float scale = 0.125f;   // 64^-0.5
#pragma unroll
  for (int mt = 0; mt < 2; ++mt) {
#pragma unroll
    for (int vr = 0; vr < 8; ++vr) {
      int row = mt * 16 + vr + hi * 8;
      int mr = (row < KW) ? row : 0;
      int c1 = 16 + cl;
      float x0 = sacc[mt][0][vr] * scale + sMask[mr * KW + cl];
      float x1 = sacc[mt][1][vr] * scale + ((c1 < KW) ? sMask[mr * KW + c1] : -1e30f);
      float mx = fmaxf(x0, x1);
#pragma unroll
      for (int o = 8; o; o >>= 1) mx = fmaxf(mx, __shfl_xor(mx, o));
      float e0 = __expf(x0 - mx), e1 = __expf(x1 - mx);
      float sm = e0 + e1;
#pragma unroll
      for (int o = 8; o; o >>= 1) sm += __shfl_xor(sm, o);
      float inv = 1.0f / sm;
      sP[wid][row * 32 + cl] = f2bf(e0 * inv);
      sP[wid][row * 32 + c1] = f2bf(e1 * inv);
    }
  }
  __syncthreads();

  // O = P @ V  (32x32 @ 32x64; single K chunk of 32)
  v8f oacc[2][4];
#pragma unroll
  for (int i = 0; i < 2; ++i)
#pragma unroll
    for (int j = 0; j < 4; ++j) oacc[i][j] = v8f_zero();
  v16bf ap[2];
#pragma unroll
  for (int mt = 0; mt < 2; ++mt)
    ap[mt] = load_frag(&sP[wid][(mt * 16 + cl) * 32], lane);
#pragma unroll
  for (int nt = 0; nt < 4; ++nt) {
    v16bf bv = load_frag_v(vp, SQ, nt * 16 + cl, lane);
#pragma unroll
    for (int mt = 0; mt < 2; ++mt)
      oacc[mt][nt] = __builtin_amdgcn_wmma_f32_16x16x32_bf16(
          false, ap[mt], false, bv, (short)0, oacc[mt][nt], false, false);
  }

  // Scatter back to natural token order (un-dilate, drop pads), bf16.
#pragma unroll
  for (int mt = 0; mt < 2; ++mt) {
#pragma unroll
    for (int vr = 0; vr < 8; ++vr) {
      int row = mt * 16 + vr + hi * 8;
      if (row < KW) {
        int n = g * 96 + row * 4 + d - PAD_N;
        if (n >= 0 && n < N_PTS) {
#pragma unroll
          for (int nt = 0; nt < 4; ++nt)
            aout[(size_t)n * C_DIM + h * HDIM + nt * 16 + cl] = f2bf(oacc[mt][nt][vr]);
        }
      }
    }
  }
}

// ---------------------------------------------------------------------------
// Kernel 3: out = attn_out(bf16) @ Wproj + bproj  (f32 result).
// ---------------------------------------------------------------------------
__global__ __launch_bounds__(256) void proj_kernel(
    const bf16_t* __restrict__ ain, const float* __restrict__ Wproj,
    const float* __restrict__ bproj, float* __restrict__ out) {
  __shared__ __align__(16) bf16_t As[128][34];
  __shared__ __align__(16) bf16_t Bs[128][34];

  const int tid = threadIdx.x, lane = tid & 31, wid = tid >> 5;
  const int wr = wid >> 1, wc = wid & 1;
  const int m0 = blockIdx.x * 128, n0 = blockIdx.y * 128;
  const int wm = wr * 32, wn = wc * 64;
  const int cl = lane & 15, hi = lane >> 4;

  v8f acc[2][4];
#pragma unroll
  for (int i = 0; i < 2; ++i)
#pragma unroll
    for (int j = 0; j < 4; ++j) acc[i][j] = v8f_zero();

  for (int kc = 0; kc < C_DIM; kc += 32) {
#pragma unroll
    for (int i = 0; i < 2; ++i) {          // A stage: 512 uint4 of bf16
      int idx = tid * 2 + i;
      int r = idx >> 2, c = idx & 3;
      union { uint4 u; bf16_t hh[8]; } cv;
      cv.u = *(const uint4*)(ain + (size_t)(m0 + r) * C_DIM + kc + c * 8);
#pragma unroll
      for (int t = 0; t < 8; ++t) As[r][c * 8 + t] = cv.hh[t];
    }
#pragma unroll
    for (int i = 0; i < 4; ++i) {          // B stage (transposed), f32 -> bf16
      int idx = tid * 4 + i;
      int kr = idx >> 5, cg = idx & 31;
      float4 vv = *(const float4*)(Wproj + (size_t)(kc + kr) * C_DIM + n0 + cg * 4);
      Bs[cg * 4 + 0][kr] = f2bf(vv.x); Bs[cg * 4 + 1][kr] = f2bf(vv.y);
      Bs[cg * 4 + 2][kr] = f2bf(vv.z); Bs[cg * 4 + 3][kr] = f2bf(vv.w);
    }
    if (kc + 32 < C_DIM)
      __builtin_prefetch(Wproj + (size_t)(kc + 32 + (tid >> 3)) * C_DIM + n0, 0, 1);
    __syncthreads();

    v16bf af[2], bfr[4];
#pragma unroll
    for (int mt = 0; mt < 2; ++mt)
      af[mt] = load_frag(&As[wm + mt * 16 + cl][0], lane);
#pragma unroll
    for (int nt = 0; nt < 4; ++nt)
      bfr[nt] = load_frag(&Bs[wn + nt * 16 + cl][0], lane);
#pragma unroll
    for (int mt = 0; mt < 2; ++mt)
#pragma unroll
      for (int nt = 0; nt < 4; ++nt)
        acc[mt][nt] = __builtin_amdgcn_wmma_f32_16x16x32_bf16(
            false, af[mt], false, bfr[nt], (short)0, acc[mt][nt], false, false);
    __syncthreads();
  }

#pragma unroll
  for (int mt = 0; mt < 2; ++mt)
#pragma unroll
    for (int vr = 0; vr < 8; ++vr) {
      int m = m0 + wm + mt * 16 + vr + hi * 8;
#pragma unroll
      for (int nt = 0; nt < 4; ++nt) {
        int col = n0 + wn + nt * 16 + cl;
        out[(size_t)m * C_DIM + col] = acc[mt][nt][vr] + bproj[col];
      }
    }
}

// ---------------------------------------------------------------------------
extern "C" void kernel_launch(void* const* d_in, const int* in_sizes, int n_in,
                              void* d_out, int out_size, void* d_ws, size_t ws_size,
                              hipStream_t stream) {
  (void)in_sizes; (void)n_in; (void)out_size; (void)ws_size;
  const float* data  = (const float*)d_in[0];
  const float* xyz   = (const float*)d_in[1];
  const float* mask  = (const float*)d_in[2];
  const float* Wqkv  = (const float*)d_in[3];
  const float* bqkv  = (const float*)d_in[4];
  const float* Wproj = (const float*)d_in[5];
  const float* bproj = (const float*)d_in[6];
  const float* rope  = (const float*)d_in[7];
  float* out = (float*)d_out;

  bf16_t* qkvw = (bf16_t*)d_ws;                                          // NA x 1536 bf16
  bf16_t* aout = (bf16_t*)((char*)d_ws + (size_t)NA_PTS * QKV_COLS * 2); // N x 512 bf16

  pad_kernel<<<192, 256, 0, stream>>>(qkvw);
  qkv_rope_kernel<<<dim3(N_PTS / 128, QKV_COLS / 128), 256, 0, stream>>>(
      data, xyz, Wqkv, bqkv, rope, qkvw);
  attn_kernel<<<dim3(NWIN, 2), 128, 0, stream>>>(qkvw, mask, aout);
  proj_kernel<<<dim3(N_PTS / 128, C_DIM / 128), 256, 0, stream>>>(
      aout, Wproj, bproj, out);
}
